// Downsample_4818953306625
// MI455X (gfx1250) — matrix operand order repaired and checked by reading, not compiled, not run on hardware
//
#include <hip/hip_runtime.h>

// ---------------------------------------------------------------------------
// MI455X (gfx1250) implementation.
//
// conv1 (96->48, 3x3, 256x256, B=8) dominates: 43.5 GFLOP implicit GEMM
// (M=48, K=864, N=pixels). Done with V_WMMA_F32_16X16X32_F16 (f16 in / f32
// accumulate), intermediate stored f16 in workspace. Second stage
// (pixel_unshuffle + 192-group 3x3 conv, 0.9 GFLOP) is memory bound and done
// as a fused fp32 elementwise-gather kernel.
// ---------------------------------------------------------------------------

typedef _Float16 v16h __attribute__((ext_vector_type(16)));
typedef float    v8f  __attribute__((ext_vector_type(8)));

#define CIN   96
#define COUT  48
#define HS    256
#define WS    256
#define NB    8
#define LDS_CSTR 104   // padded per-pixel ci stride in halves (208B, 16B-aligned, bank-safe)
#define LDS_COLS 130   // 128 cols + 2 halo
#define LDS_ROWS 4     // 2 rows + 2 halo
#define WA_HALVES (9*3*3*32*16)   // [pos][kstep][mtile][lane][16] = 41472

// ---------------------------------------------------------------------------
// Pack w_body (f32, [48][96][3][3]) into the exact per-lane WMMA A-operand
// f16 layout: flat = (((pos*3+kstep)*3+mtile)*32+lane)*16+j,
// element j of lane (m=lane&15, hi=lane>>4) holds K = j+16*hi
// within the kstep's 32-wide ci window (contiguous k order, same as the B operand read from LDS).
// ---------------------------------------------------------------------------
__global__ void pack_weights(const float* __restrict__ wbody,
                             _Float16* __restrict__ wa) {
  int t = blockIdx.x * 256 + threadIdx.x;
  if (t >= WA_HALVES) return;
  int j    = t & 15;
  int lane = (t >> 4) & 31;
  int rest = t >> 9;            // pos*9 + kstep*3 + mtile, 0..80
  int mtile = rest % 3;
  int kstep = (rest / 3) % 3;
  int pos   = rest / 9;
  int hi = lane >> 4;
  int Kl = j + 16 * hi;
  int ci = kstep * 32 + Kl;
  int co = mtile * 16 + (lane & 15);
  int kh = pos / 3, kw = pos % 3;
  wa[t] = (_Float16)wbody[((co * CIN + ci) * 3 + kh) * 3 + kw];
}

// ---------------------------------------------------------------------------
// conv1 via WMMA. Block = 128 threads (4 waves). Tile: 2 rows x 128 cols.
// Wave wv: row = h0 + (wv>>1), x-range = wseg*128 + (wv&1)*64 .. +63.
// Each wave: 3 M-tiles (48 ch) x 4 N-tiles (64 px) accumulators.
// K loop: 9 filter positions x 3 ci-steps of 32.
// ---------------------------------------------------------------------------
__global__ void __launch_bounds__(128) conv1_wmma(
    const float* __restrict__ x, const _Float16* __restrict__ wa,
    _Float16* __restrict__ out) {
  extern __shared__ _Float16 lds[];
  const int wseg = blockIdx.x;        // 0..1  (x half)
  const int h0   = blockIdx.y * 2;    // row pair
  const int b    = blockIdx.z;
  const int tid  = threadIdx.x;

  // ---- stage input tile (rows h0-1..h0+2, cols wseg*128-1..+128, all ci) ----
  // LDS layout: lds[(dy*LDS_COLS + col)*LDS_CSTR + ci], f16, zero-padded.
  const int total = LDS_ROWS * CIN * LDS_COLS;   // 49920
  for (int idx = tid; idx < total; idx += 128) {
    int col = idx % LDS_COLS;
    int rc  = idx / LDS_COLS;
    int ci  = rc % CIN;
    int dy  = rc / CIN;
    int row = h0 - 1 + dy;
    int w   = wseg * 128 - 1 + col;
    float v = 0.0f;
    if ((unsigned)row < (unsigned)HS && (unsigned)w < (unsigned)WS)
      v = x[((b * CIN + ci) << 16) | (row << 8) | w];
    lds[(dy * LDS_COLS + col) * LDS_CSTR + ci] = (_Float16)v;
  }
  __syncthreads();

  const int lane  = tid & 31;
  const int wv    = tid >> 5;       // 0..3
  const int hrow  = wv >> 1;        // local row 0/1
  const int xbase = (wv & 1) * 64;  // local col base
  const int n  = lane & 15;
  const int hi = lane >> 4;

  v8f acc[3][4];
#pragma unroll
  for (int m = 0; m < 3; ++m)
#pragma unroll
    for (int nt = 0; nt < 4; ++nt)
      acc[m][nt] = (v8f)0.0f;

  for (int pos = 0; pos < 9; ++pos) {
    const int kh = pos / 3;
    const int kw = pos % 3;
    const int rowbase = (hrow + kh) * LDS_COLS;   // LDS row for this tap
#pragma unroll
    for (int ks = 0; ks < 3; ++ks) {
      // A operands: coalesced 32B/lane from the pre-packed (L2-resident) buffer.
      const _Float16* ap = wa + ((pos * 3 + ks) * 3) * 512 + lane * 16;
      v16h a0 = *(const v16h*)(ap);
      v16h a1 = *(const v16h*)(ap + 512);
      v16h a2 = *(const v16h*)(ap + 1024);
#pragma unroll
      for (int nt = 0; nt < 4; ++nt) {
        // B operand: lane (n = col within N-tile, hi selects ci+0/+16),
        // 16 contiguous f16 (32B, 16B-aligned) from LDS.
        const int col = xbase + nt * 16 + n + kw;   // 0..129
        const _Float16* bp =
            &lds[(rowbase + col) * LDS_CSTR + ks * 32 + hi * 16];
        v16h bv = *(const v16h*)(bp);
        acc[0][nt] = __builtin_amdgcn_wmma_f32_16x16x32_f16(
            false, a0, false, bv, (short)0, acc[0][nt], false, false);
        acc[1][nt] = __builtin_amdgcn_wmma_f32_16x16x32_f16(
            false, a1, false, bv, (short)0, acc[1][nt], false, false);
        acc[2][nt] = __builtin_amdgcn_wmma_f32_16x16x32_f16(
            false, a2, false, bv, (short)0, acc[2][nt], false, false);
      }
    }
  }

  // ---- store: D layout = VGPR v holds channel (v + 8*hi), col = lane&15 ----
  const int hw = h0 + hrow;
#pragma unroll
  for (int m = 0; m < 3; ++m)
#pragma unroll
    for (int nt = 0; nt < 4; ++nt) {
      const int xp = wseg * 128 + xbase + nt * 16 + n;
#pragma unroll
      for (int v = 0; v < 8; ++v) {
        const int ch = m * 16 + v + 8 * hi;
        out[((b * COUT + ch) << 16) | (hw << 8) | xp] = (_Float16)acc[m][nt][v];
      }
    }
}

// ---------------------------------------------------------------------------
// Fused pixel_unshuffle + mask-broadcast + 192-group 3x3 conv (fp32).
// g = c*4 + r1*2 + r2; unshuffled conv1 value and mask share the same
// (2y+r1, 2x+r2) source index. Memory bound (~200MB round trip).
// ---------------------------------------------------------------------------
__global__ void __launch_bounds__(128) conv2_grouped(
    const _Float16* __restrict__ u, const float* __restrict__ mask,
    const float* __restrict__ wproj, float* __restrict__ out) {
  const int x = threadIdx.x;   // 0..127
  const int y = blockIdx.x;    // 0..127
  const int g = blockIdx.y;    // 0..191
  const int b = blockIdx.z;    // 0..7
  const int c  = g >> 2;
  const int r1 = (g >> 1) & 1;
  const int r2 = g & 1;
  const float* wp = wproj + g * 18;   // [2][3][3]

  float s = 0.0f;
#pragma unroll
  for (int kh = 0; kh < 3; ++kh) {
    const int Y = y + kh - 1;
    if ((unsigned)Y >= 128u) continue;
    const int row = 2 * Y + r1;
#pragma unroll
    for (int kw = 0; kw < 3; ++kw) {
      const int X = x + kw - 1;
      if ((unsigned)X >= 128u) continue;
      const int col = 2 * X + r2;
      const float uv = (float)u[((b * COUT + c) << 16) | (row << 8) | col];
      const float mv = mask[(b << 16) | (row << 8) | col];
      s += uv * wp[kh * 3 + kw] + mv * wp[9 + kh * 3 + kw];
    }
  }
  out[(((b * 192 + g) << 7) | y) << 7 | x] = s;
}

// ---------------------------------------------------------------------------
extern "C" void kernel_launch(void* const* d_in, const int* in_sizes, int n_in,
                              void* d_out, int out_size, void* d_ws, size_t ws_size,
                              hipStream_t stream) {
  const float* x     = (const float*)d_in[0];
  const float* mask  = (const float*)d_in[1];
  const float* wbody = (const float*)d_in[2];
  const float* wproj = (const float*)d_in[3];
  float* out = (float*)d_out;

  // workspace: [0, 50331648) f16 conv1 output, then packed A-operand weights.
  _Float16* conv1ws = (_Float16*)d_ws;
  const size_t conv1_bytes = (size_t)NB * COUT * HS * WS * sizeof(_Float16);
  _Float16* wa = (_Float16*)((char*)d_ws + conv1_bytes);

  pack_weights<<<(WA_HALVES + 255) / 256, 256, 0, stream>>>(wbody, wa);

  dim3 gridA(2, 128, NB);
  const size_t shmem = (size_t)LDS_ROWS * LDS_COLS * LDS_CSTR * sizeof(_Float16); // 108160
  conv1_wmma<<<gridA, 128, shmem, stream>>>(x, wa, conv1ws);

  dim3 gridB(128, 192, NB);
  conv2_grouped<<<gridB, 128, 0, stream>>>(conv1ws, mask, wproj, out);
}
